// PositionEncoder_42279658061914
// MI455X (gfx1250) — matrix operand order repaired and checked by reading, not compiled, not run on hardware
//
#include <hip/hip_runtime.h>
#include <hip/hip_bf16.h>

typedef __attribute__((ext_vector_type(16))) _Float16 v16h;
typedef __attribute__((ext_vector_type(8)))  _Float16 v8h;
typedef __attribute__((ext_vector_type(8)))  float    v8f;

#define N_NODES 4096
#define N_DIM   128
#define E_DIM   64
#define UEV_DIM (2 * N_DIM + E_DIM)   // 320 = 10 chunks of K=32

// ---------------------------------------------------------------------------
// Kernel 1: zero the [N,N] f32 output (harness poisons it with 0xAA).
// 64 MB of pure stores -> HBM-bound, ~3us at 23.3 TB/s.
// ---------------------------------------------------------------------------
__global__ __launch_bounds__(256) void pe_zero_kernel(float4* __restrict__ out4,
                                                      float* __restrict__ out,
                                                      int n4, int n) {
    int i = blockIdx.x * blockDim.x + threadIdx.x;
    if (i < n4) out4[i] = make_float4(0.f, 0.f, 0.f, 0.f);
    if (i == 0) {
        for (int r = n4 * 4; r < n; ++r) out[r] = 0.f;  // tail (empty for 16M)
    }
}

// ---------------------------------------------------------------------------
// Kernel 2: per-wave, score 16 edges with v_wmma_f32_16x16x32_f16, then
// sigmoid and atomic-scatter into the output.
//
//   A (16x32 f16)  = W_e chunk broadcast to all 16 rows
//   B (32x16 f16)  = column n = edge n's concat(v[u], e, v[v]) chunk
//   D (16x16 f32)  = identical rows; row 0 col n = raw score of edge n
//
// A per-lane layout (16-bit A, 16x32): lanes 0-15 hold K=k0..k0+15,
// lanes 16-31 hold K=k0+16..k0+31, consecutive and pair-packed in VGPR order.
// B per-lane layout: lane n holds column n with K=k0..k0+15 (lanes 0-15) or
// K=k0+16..k0+31 (lanes 16-31), consecutive and pair-packed.
// D layout: VGPR0 lanes 0-15 = (M=0, N=lane)  -> acc[0] is the score.
// ---------------------------------------------------------------------------
__global__ __launch_bounds__(256) void pe_edge_scatter_kernel(
    const float* __restrict__ vfeat,   // [N_NODES, N_DIM]
    const float* __restrict__ efeat,   // [N_EDGES, E_DIM]
    const float* __restrict__ We,      // [UEV_DIM]
    const float* __restrict__ be,      // [1]
    const int*   __restrict__ us,      // [N_EDGES]
    const int*   __restrict__ vs,      // [N_EDGES]
    float* __restrict__ out,           // [N_NODES, N_NODES]
    int E)
{
    __shared__ _Float16 sW[UEV_DIM];
    const int tid = threadIdx.x;
    for (int i = tid; i < UEV_DIM; i += blockDim.x)
        sW[i] = (_Float16)We[i];
    __syncthreads();

    const int lane   = tid & 31;
    const int tile   = (blockIdx.x * blockDim.x + tid) >> 5;  // 16 edges/tile
    if (tile * 16 >= E) return;                               // uniform per wave

    const int  e_local = lane & 15;
    const int  laneHi  = lane >> 4;                           // 0 or 1
    const int  e       = tile * 16 + e_local;
    const bool valid   = (e < E);
    const int  ec      = valid ? e : (E - 1);                 // clamp loads only

    const int u = us[ec];
    const int v = vs[ec];

    v8f acc = {};
    #pragma unroll
    for (int c = 0; c < UEV_DIM / 32; ++c) {
        const int k0 = c * 32;

        // ---- A: W_e chunk, broadcast across the 16 rows -------------------
        const int abase = k0 + laneHi * 16;                   // 8-half aligned
        v8h alo = *(const v8h*)&sW[abase];                    // K = abase..+7
        v8h ahi = *(const v8h*)&sW[abase + 8];                // K = abase+8..+15
        v16h a = __builtin_shufflevector(alo, ahi,
                 0, 1, 2, 3, 4, 5, 6, 7, 8, 9, 10, 11, 12, 13, 14, 15);

        // ---- B: 16 consecutive f32 of this edge's virtual uev row ---------
        // Segment boundaries (128, 192) are multiples of 32, so the whole
        // chunk lives in one source tensor; k0 is constant after unroll.
        const float* srcbase;
        int roff;
        if (k0 < N_DIM)              { srcbase = vfeat + (size_t)u  * N_DIM; roff = k0; }
        else if (k0 < N_DIM + E_DIM) { srcbase = efeat + (size_t)ec * E_DIM; roff = k0 - N_DIM; }
        else                         { srcbase = vfeat + (size_t)v  * N_DIM; roff = k0 - N_DIM - E_DIM; }

        const float4* s4 = (const float4*)(srcbase + roff + laneHi * 16);
        float4 f0 = s4[0], f1 = s4[1], f2 = s4[2], f3 = s4[3];
        v16h b;
        b[0]  = (_Float16)f0.x; b[1]  = (_Float16)f0.y;
        b[2]  = (_Float16)f0.z; b[3]  = (_Float16)f0.w;
        b[4]  = (_Float16)f1.x; b[5]  = (_Float16)f1.y;
        b[6]  = (_Float16)f1.z; b[7]  = (_Float16)f1.w;
        b[8]  = (_Float16)f2.x; b[9]  = (_Float16)f2.y;
        b[10] = (_Float16)f2.z; b[11] = (_Float16)f2.w;
        b[12] = (_Float16)f3.x; b[13] = (_Float16)f3.y;
        b[14] = (_Float16)f3.z; b[15] = (_Float16)f3.w;

        acc = __builtin_amdgcn_wmma_f32_16x16x32_f16(
                  /*neg_a=*/false, a, /*neg_b=*/false, b,
                  /*c_mod=*/(short)0, acc, /*reuse_a=*/false, /*reuse_b=*/false);
    }

    // ---- sigmoid + sparse A.diag(w).A^T scatter (4 rank-1 taps per edge) --
    if (lane < 16 && valid) {
        const float x = acc[0] + be[0];
        const float w = 1.0f / (1.0f + __expf(-x));
        atomicAdd(&out[(size_t)u * N_NODES + u], w);
        if (u != v) {
            atomicAdd(&out[(size_t)u * N_NODES + v], w);
            atomicAdd(&out[(size_t)v * N_NODES + u], w);
            atomicAdd(&out[(size_t)v * N_NODES + v], w);
        }
    }
}

// ---------------------------------------------------------------------------
extern "C" void kernel_launch(void* const* d_in, const int* in_sizes, int n_in,
                              void* d_out, int out_size, void* d_ws, size_t ws_size,
                              hipStream_t stream) {
    const float* vfeat = (const float*)d_in[0];   // [4096,128] f32
    const float* efeat = (const float*)d_in[1];   // [16384,64] f32
    const float* We    = (const float*)d_in[2];   // [320,1]    f32
    const float* be    = (const float*)d_in[3];   // [1]        f32
    const int*   us    = (const int*)  d_in[4];   // [16384]    i32
    const int*   vs    = (const int*)  d_in[5];   // [16384]    i32
    // d_in[6] node_edge_matrix is intentionally unused: the incidence matrix
    // has exactly the endpoints us/vs as nonzeros, so A.diag(w).A^T is a
    // 4-tap scatter per edge. Skipping it avoids reading 256 MB.
    float* out = (float*)d_out;

    const int E  = in_sizes[4];
    const int n  = out_size;          // 4096*4096
    const int n4 = n / 4;

    pe_zero_kernel<<<(n4 + 255) / 256, 256, 0, stream>>>((float4*)out, out, n4, n);

    const int waves  = (E + 15) / 16;         // one wave per 16-edge tile
    const int blocks = (waves + 7) / 8;       // 8 waves (256 threads) per block
    pe_edge_scatter_kernel<<<blocks, 256, 0, stream>>>(
        vfeat, efeat, We, be, us, vs, out, E);
}
